// PaiNNModel_25898652795341
// MI455X (gfx1250) — compile-verified, matrix operand order checked by reading, not run on hardware
//
#include <hip/hip_runtime.h>
#include <hip/hip_bf16.h>

#define UNITS 256
#define NUM_RADIAL 20
#define DEPTH 6
#define N_NODES 10000
#define N_EDGES 160000
#define NUM_TARGETS_C 128
#define EPS_F 1.1920929e-7f

typedef __attribute__((ext_vector_type(16))) __bf16 v16bf;
typedef __attribute__((ext_vector_type(8)))  float  v8f;

__device__ __forceinline__ float silu_fast(float x)
{
  // x * sigmoid(x); v_rcp_f32 instead of IEEE divide
  return x * __builtin_amdgcn_rcpf(1.f + __expf(-x));
}

// ---------------------------------------------------------------------------
// WMMA GEMM: C[M,Nout] = act(A[M,K] @ Wt[Nout,K]^T + bias)
// A, Wt row-major f32; converted to bf16 on the fly.
// Block = 128 threads (4 waves). All 4 waves share the same 16 A rows; each
// wave owns 4 consecutive 16-col tiles -> block tile = 16 rows x 256 cols.
// One A fragment feeds 4 back-to-back WMMAs (4 accumulators, 4x A reuse).
// Requires M%16==0, Nout%256==0, K%32==0.
//
// Fragment layout per CDNA5 ISA 7.12.2 (wave32):
//  A 16x32 bf16: lane L (half=L/16, m=L%16) holds row m;
//    elems 0..7  -> K = half*8 + (0..7);  elems 8..15 -> K = half*8+16+(0..7)
//  B 32x16 bf16: lane L holds column n=L%16, same K packing; for row-major
//    Wt[Nout,K] this is the identical per-lane pattern on row (col).
//  D (v8f): elem r -> C[row0 + r + 8*half][coltile + L%16]
// ---------------------------------------------------------------------------
template<int ACT>
__global__ __launch_bounds__(128)
void k_gemm_wmma(const float* __restrict__ A, const float* __restrict__ Wt,
                 const float* __restrict__ bias, float* __restrict__ C,
                 int M, int Nout, int K)
{
  const int lane = threadIdx.x & 31;
  const int wave = threadIdx.x >> 5;
  const int half = lane >> 4;
  const int l16  = lane & 15;

  const int arow = blockIdx.y * 16 + l16;
  const int cbase = blockIdx.x * 256 + wave * 64;        // wave's first column

  const float* ap0 = A + (size_t)arow * K + half * 8;
  const float* bp0 = Wt + (size_t)(cbase + l16) * K + half * 8;
  const size_t bstride = (size_t)16 * K;                 // 16 cols per tile

  v8f acc0 = {}, acc1 = {}, acc2 = {}, acc3 = {};
  for (int kk = 0; kk < K; kk += 32) {
    const float* ap = ap0 + kk;
    __builtin_prefetch(ap + 32, 0, 1);                   // next A K-slice
    v16bf af;
#pragma unroll
    for (int e = 0; e < 8; ++e) {
      af[e]     = (__bf16)ap[e];
      af[e + 8] = (__bf16)ap[e + 16];
    }
    v16bf bf0, bf1, bf2, bf3;
    {
      const float* bp = bp0 + kk;
#pragma unroll
      for (int e = 0; e < 8; ++e) { bf0[e] = (__bf16)bp[e]; bf0[e+8] = (__bf16)bp[e+16]; }
      bp += bstride;
#pragma unroll
      for (int e = 0; e < 8; ++e) { bf1[e] = (__bf16)bp[e]; bf1[e+8] = (__bf16)bp[e+16]; }
      bp += bstride;
#pragma unroll
      for (int e = 0; e < 8; ++e) { bf2[e] = (__bf16)bp[e]; bf2[e+8] = (__bf16)bp[e+16]; }
      bp += bstride;
#pragma unroll
      for (int e = 0; e < 8; ++e) { bf3[e] = (__bf16)bp[e]; bf3[e+8] = (__bf16)bp[e+16]; }
    }
    acc0 = __builtin_amdgcn_wmma_f32_16x16x32_bf16(false, af, false, bf0, (short)0, acc0, false, false);
    acc1 = __builtin_amdgcn_wmma_f32_16x16x32_bf16(false, af, false, bf1, (short)0, acc1, false, false);
    acc2 = __builtin_amdgcn_wmma_f32_16x16x32_bf16(false, af, false, bf2, (short)0, acc2, false, false);
    acc3 = __builtin_amdgcn_wmma_f32_16x16x32_bf16(false, af, false, bf3, (short)0, acc3, false, false);
  }

  const int orow0 = blockIdx.y * 16 + half * 8;
  v8f accs[4] = {acc0, acc1, acc2, acc3};
#pragma unroll
  for (int t = 0; t < 4; ++t) {
    const int ocol = cbase + t * 16 + l16;
    const float bv = bias ? bias[ocol] : 0.f;
#pragma unroll
    for (int r = 0; r < 8; ++r) {
      float x = accs[t][r] + bv;
      if (ACT == 1) x = silu_fast(x);
      C[(size_t)(orow0 + r) * Nout + ocol] = x;
    }
  }
}

// ---------------------------------------------------------------------------
// Fused per-edge kernel: radial basis + envelope, w = rbf @ w_w^T + w_b,
// sw = s3[j] * w, scatter-add ds into dz[i], dv into dv[i].
// One 256-thread block per edge; thread f handles features (f, 256+f, 512+f).
// ---------------------------------------------------------------------------
__global__ __launch_bounds__(256)
void k_edge(const float* __restrict__ pos, const int* __restrict__ edges,
            const float* __restrict__ v, const float* __restrict__ s3,
            const float* __restrict__ freq, const float* __restrict__ w_w,
            const float* __restrict__ w_b, float* __restrict__ dz,
            float* __restrict__ dv)
{
  const int e = blockIdx.x;
  const int f = threadIdx.x;
  const int i = edges[2 * e + 0];
  const int j = edges[2 * e + 1];

  __shared__ float sh_rijn[3];
  __shared__ float sh_rbf[NUM_RADIAL];
  __shared__ float sh_x, sh_dcut;

  if (f == 0) {
    float rx = pos[3 * i + 0] - pos[3 * j + 0];
    float ry = pos[3 * i + 1] - pos[3 * j + 1];
    float rz = pos[3 * i + 2] - pos[3 * j + 2];
    float dist = sqrtf(rx * rx + ry * ry + rz * rz);
    bool  m    = dist > EPS_F;
    float safe = m ? dist : 1.f;
    float inv  = m ? 1.f / safe : 0.f;
    sh_rijn[0] = rx * inv; sh_rijn[1] = ry * inv; sh_rijn[2] = rz * inv;
    float x  = safe * 0.2f;                       // 1/CUTOFF
    float x2 = x * x, x4 = x2 * x2, x5 = x4 * x, x6 = x5 * x, x7 = x6 * x;
    float env  = 1.f / x - 28.f * x5 + 48.f * x6 - 21.f * x7;  // p=6 coeffs
    float dcut = (x < 1.f) ? env : 0.f;
    sh_x = x;
    sh_dcut = m ? dcut : 0.f;                     // fold edge mask into d_cut
  }
  __syncthreads();
  if (f < NUM_RADIAL) sh_rbf[f] = sh_dcut * __sinf(freq[f] * sh_x);
  __syncthreads();

  // w[f], w[256+f], w[512+f] : tiny K=20 dot products (weights L2-resident)
  float wv[3];
#pragma unroll
  for (int p = 0; p < 3; ++p) {
    int o = p * UNITS + f;
    float acc = w_b[o];
    const float* wr = w_w + (size_t)o * NUM_RADIAL;
#pragma unroll
    for (int r = 0; r < NUM_RADIAL; ++r) acc += sh_rbf[r] * wr[r];
    wv[p] = acc;
  }

  const float* sj = s3 + (size_t)j * (3 * UNITS);
  float sw1 = sj[f]             * wv[0];
  float sw2 = sj[UNITS + f]     * wv[1];
  float sw3 = sj[2 * UNITS + f] * wv[2];

  atomicAdd(&dz[(size_t)i * UNITS + f], sw1);
#pragma unroll
  for (int d = 0; d < 3; ++d) {
    float val = v[((size_t)j * 3 + d) * UNITS + f] * sw2 + sh_rijn[d] * sw3;
    atomicAdd(&dv[((size_t)i * 3 + d) * UNITS + f], val);
  }
}

// ------------------------- small elementwise kernels ------------------------
__global__ __launch_bounds__(256)
void k_embed(const int* __restrict__ attr, const float* __restrict__ embed,
             float* __restrict__ z)
{
  int idx = blockIdx.x * 256 + threadIdx.x;       // N*256 threads
  int n = idx >> 8, f = idx & 255;
  z[idx] = embed[(size_t)attr[n] * UNITS + f];
}

__global__ __launch_bounds__(256)
void k_fill(float* __restrict__ p, float val, int count)
{
  int idx = blockIdx.x * 256 + threadIdx.x;
  if (idx < count) p[idx] = val;
}

__global__ __launch_bounds__(256)
void k_axpy(float* __restrict__ y, const float* __restrict__ x, int count)
{
  int idx = blockIdx.x * 256 + threadIdx.x;
  if (idx < count) y[idx] += x[idx];
}

__global__ __launch_bounds__(256)
void k_vprod_norm(const float* __restrict__ vu, const float* __restrict__ vv,
                  float* __restrict__ vprod, float* __restrict__ vnorm)
{
  int idx = blockIdx.x * 256 + threadIdx.x;       // N*256 threads
  int n = idx >> 8, f = idx & 255;
  float pv = 0.f, nv = 0.f;
#pragma unroll
  for (int d = 0; d < 3; ++d) {
    float a = vu[((size_t)n * 3 + d) * UNITS + f];
    float b = vv[((size_t)n * 3 + d) * UNITS + f];
    pv += a * b;
    nv += b * b;
  }
  vprod[idx] = pv;
  vnorm[idx] = sqrtf(nv);
}

__global__ __launch_bounds__(256)
void k_cat(const float* __restrict__ z, const float* __restrict__ vnorm,
           float* __restrict__ cat)
{
  int idx = blockIdx.x * 256 + threadIdx.x;       // N*256 threads
  int n = idx >> 8, f = idx & 255;
  cat[(size_t)n * 512 + f]       = z[idx];
  cat[(size_t)n * 512 + 256 + f] = vnorm[idx];
}

__global__ __launch_bounds__(256)
void k_update(float* __restrict__ z, float* __restrict__ v,
              const float* __restrict__ a, const float* __restrict__ vprod,
              const float* __restrict__ vu)
{
  int idx = blockIdx.x * 256 + threadIdx.x;       // N*256 threads
  int n = idx >> 8, f = idx & 255;
  const float* an = a + (size_t)n * (3 * UNITS);
  z[idx] += vprod[idx] * an[UNITS + f] + an[2 * UNITS + f];   // a_sv, a_ss
  float avv = an[f];                                           // a_vv
#pragma unroll
  for (int d = 0; d < 3; ++d) {
    size_t o = ((size_t)n * 3 + d) * UNITS + f;
    v[o] += vu[o] * avv;
  }
}

__global__ __launch_bounds__(256)
void k_pool(const float* __restrict__ z, const int* __restrict__ atom,
            float* __restrict__ pooled, float* __restrict__ counts)
{
  int idx = blockIdx.x * 256 + threadIdx.x;       // N*256 threads
  int n = idx >> 8, f = idx & 255;
  int t = atom[n];
  atomicAdd(&pooled[(size_t)t * UNITS + f], z[idx]);
  if (f == 0) atomicAdd(&counts[t], 1.f);
}

__global__ __launch_bounds__(256)
void k_pool_div(float* __restrict__ pooled, const float* __restrict__ counts)
{
  int idx = blockIdx.x * 256 + threadIdx.x;       // 128*256 threads
  pooled[idx] /= fmaxf(counts[idx >> 8], 1.f);
}

__global__ __launch_bounds__(128)
void k_head_out(const float* __restrict__ h, const float* __restrict__ w2,
                const float* __restrict__ b2, float* __restrict__ out)
{
  int t = threadIdx.x;                            // 128 threads
  float acc = b2[0];
  const float* hr = h + (size_t)t * UNITS;
  for (int f = 0; f < UNITS; ++f) acc += hr[f] * w2[f];
  out[t] = acc;
}

// ---------------------------------------------------------------------------
static inline float* ws_alloc(char*& p, size_t nfloat)
{
  float* r = (float*)p;
  p += ((nfloat * sizeof(float) + 255) & ~(size_t)255);
  return r;
}

extern "C" void kernel_launch(void* const* d_in, const int* in_sizes, int n_in,
                              void* d_out, int out_size, void* d_ws, size_t ws_size,
                              hipStream_t stream)
{
  (void)in_sizes; (void)n_in; (void)out_size; (void)ws_size;

  // Inputs in setup_inputs() dict order (params dict flattened in order)
  const int*   attr   = (const int*)  d_in[0];
  const float* pos    = (const float*)d_in[1];
  const int*   edges  = (const int*)  d_in[2];
  const int*   atom   = (const int*)  d_in[3];
  /* d_in[4] = num_targets scalar (compile-time 128) */
  const float* freq   = (const float*)d_in[5];
  const float* embed  = (const float*)d_in[6];
  const float* s_w    = (const float*)d_in[7];
  const float* s_b    = (const float*)d_in[8];
  const float* phi_w  = (const float*)d_in[9];
  const float* phi_b  = (const float*)d_in[10];
  const float* w_w    = (const float*)d_in[11];
  const float* w_b    = (const float*)d_in[12];
  const float* u1_w   = (const float*)d_in[13];
  const float* u1_b   = (const float*)d_in[14];
  const float* lin_u  = (const float*)d_in[15];
  const float* lin_v  = (const float*)d_in[16];
  const float* a_w    = (const float*)d_in[17];
  const float* a_b    = (const float*)d_in[18];
  const float* mlp1_w = (const float*)d_in[19];
  const float* mlp1_b = (const float*)d_in[20];
  const float* mlp2_w = (const float*)d_in[21];
  const float* mlp2_b = (const float*)d_in[22];
  float* out = (float*)d_out;

  const int N = N_NODES, E = N_EDGES;
  const size_t NF  = (size_t)N * UNITS;       // 2.56M
  const size_t NV  = (size_t)N * 3 * UNITS;   // 7.68M
  const size_t NC  = (size_t)N * 2 * UNITS;   // 5.12M

  char* wp = (char*)d_ws;
  float* z     = ws_alloc(wp, NF);
  float* v     = ws_alloc(wp, NV);
  float* s1    = ws_alloc(wp, NF);
  float* s3    = ws_alloc(wp, (size_t)N * 3 * UNITS);   // also reused for 'a'
  float* dz    = ws_alloc(wp, NF);
  float* dv    = ws_alloc(wp, NV);
  float* vv    = ws_alloc(wp, NV);
  float* vu    = ws_alloc(wp, NV);
  float* vprod = ws_alloc(wp, NF);
  float* vnorm = ws_alloc(wp, NF);
  float* cat   = ws_alloc(wp, NC);
  float* pooled= ws_alloc(wp, (size_t)NUM_TARGETS_C * UNITS);
  float* counts= ws_alloc(wp, NUM_TARGETS_C);
  float* h     = ws_alloc(wp, (size_t)NUM_TARGETS_C * UNITS);

  const int bNF = (int)(NF / 256);   // blocks for N*256 elementwise
  const int bNV = (int)(NV / 256);

  // z = embed[attr]; v = eps
  k_embed<<<bNF, 256, 0, stream>>>(attr, embed, z);
  k_fill<<<bNV, 256, 0, stream>>>(v, EPS_F, (int)NV);

  for (int l = 0; l < DEPTH; ++l) {
    const float* s_w_l   = s_w   + (size_t)l * UNITS * UNITS;
    const float* s_b_l   = s_b   + (size_t)l * UNITS;
    const float* phi_w_l = phi_w + (size_t)l * 3 * UNITS * UNITS;
    const float* phi_b_l = phi_b + (size_t)l * 3 * UNITS;
    const float* w_w_l   = w_w   + (size_t)l * 3 * UNITS * NUM_RADIAL;
    const float* w_b_l   = w_b   + (size_t)l * 3 * UNITS;
    const float* u1_w_l  = u1_w  + (size_t)l * UNITS * 2 * UNITS;
    const float* u1_b_l  = u1_b  + (size_t)l * UNITS;
    const float* lin_u_l = lin_u + (size_t)l * UNITS * UNITS;
    const float* lin_v_l = lin_v + (size_t)l * UNITS * UNITS;
    const float* a_w_l   = a_w   + (size_t)l * 3 * UNITS * UNITS;
    const float* a_b_l   = a_b   + (size_t)l * 3 * UNITS;

    // s1 = silu(z @ s_w^T + s_b)          [N,256] x [256,256]
    k_gemm_wmma<1><<<dim3(UNITS / 256, N / 16), 128, 0, stream>>>(
        z, s_w_l, s_b_l, s1, N, UNITS, UNITS);
    // s3 = s1 @ phi_w^T + phi_b           [N,256] x [768,256]
    k_gemm_wmma<0><<<dim3(3 * UNITS / 256, N / 16), 128, 0, stream>>>(
        s1, phi_w_l, phi_b_l, s3, N, 3 * UNITS, UNITS);

    // message passing (fused rbf + w-GEMM + gather + scatter-add)
    hipMemsetAsync(dz, 0, NF * sizeof(float), stream);
    hipMemsetAsync(dv, 0, NV * sizeof(float), stream);
    k_edge<<<E, 256, 0, stream>>>(pos, edges, v, s3, freq, w_w_l, w_b_l, dz, dv);
    k_axpy<<<bNF, 256, 0, stream>>>(z, dz, (int)NF);
    k_axpy<<<bNV, 256, 0, stream>>>(v, dv, (int)NV);

    // vv = v @ lin_v^T ; vu = v @ lin_u^T     [3N,256] x [256,256]
    k_gemm_wmma<0><<<dim3(UNITS / 256, 3 * N / 16), 128, 0, stream>>>(
        v, lin_v_l, nullptr, vv, 3 * N, UNITS, UNITS);
    k_gemm_wmma<0><<<dim3(UNITS / 256, 3 * N / 16), 128, 0, stream>>>(
        v, lin_u_l, nullptr, vu, 3 * N, UNITS, UNITS);
    k_vprod_norm<<<bNF, 256, 0, stream>>>(vu, vv, vprod, vnorm);

    // a = silu([z,vnorm] @ u1_w^T + u1_b) @ a_w^T + a_b
    k_cat<<<bNF, 256, 0, stream>>>(z, vnorm, cat);
    k_gemm_wmma<1><<<dim3(UNITS / 256, N / 16), 128, 0, stream>>>(
        cat, u1_w_l, u1_b_l, s1, N, UNITS, 2 * UNITS);
    k_gemm_wmma<0><<<dim3(3 * UNITS / 256, N / 16), 128, 0, stream>>>(
        s1, a_w_l, a_b_l, s3, N, 3 * UNITS, UNITS);

    k_update<<<bNF, 256, 0, stream>>>(z, v, s3, vprod, vu);
  }

  // mean-pool by atom_index, then MLP head
  hipMemsetAsync(pooled, 0, (size_t)NUM_TARGETS_C * UNITS * sizeof(float), stream);
  hipMemsetAsync(counts, 0, NUM_TARGETS_C * sizeof(float), stream);
  k_pool<<<bNF, 256, 0, stream>>>(z, atom, pooled, counts);
  k_pool_div<<<NUM_TARGETS_C, 256, 0, stream>>>(pooled, counts);
  k_gemm_wmma<1><<<dim3(UNITS / 256, NUM_TARGETS_C / 16), 128, 0, stream>>>(
      pooled, mlp1_w, mlp1_b, h, NUM_TARGETS_C, UNITS, UNITS);
  k_head_out<<<1, NUM_TARGETS_C, 0, stream>>>(h, mlp2_w, mlp2_b, out);
}
